// QuantLinear_335007449411
// MI455X (gfx1250) — 8-bit matrix operand order repaired and checked by reading, compiled, not run on hardware
//
#include <hip/hip_runtime.h>
#include <cstdint>

typedef __attribute__((ext_vector_type(8))) int v8i;

#define WAVE 32

__global__ __launch_bounds__(256) void rowquant_kernel(
    const float* __restrict__ src, uint8_t* __restrict__ dst,
    float4* __restrict__ param, int K, float qmax) {
  const int row  = blockIdx.x;
  const int tid  = threadIdx.x;
  const int lane = tid & (WAVE - 1);
  const int wave = tid >> 5;
  const int nwaves = blockDim.x >> 5;
  const float* s = src + (size_t)row * K;

  float vmin =  INFINITY;
  float vmax = -INFINITY;
  for (int i = tid; i < K; i += blockDim.x) {
    float v = s[i];
    vmin = fminf(vmin, v);
    vmax = fmaxf(vmax, v);
  }
  for (int o = 16; o > 0; o >>= 1) {
    vmin = fminf(vmin, __shfl_xor(vmin, o, WAVE));
    vmax = fmaxf(vmax, __shfl_xor(vmax, o, WAVE));
  }
  __shared__ float smin[8], smax[8];
  __shared__ int   ssum[8];
  if (lane == 0) { smin[wave] = vmin; smax[wave] = vmax; }
  __syncthreads();
  vmin = smin[0]; vmax = smax[0];
  for (int w = 1; w < nwaves; ++w) {
    vmin = fminf(vmin, smin[w]);
    vmax = fmaxf(vmax, smax[w]);
  }

  const float scale = fmaxf(vmax - vmin, 1e-5f) / qmax;
  const float inv_s = 1.0f / scale;
  const float zp    = rintf(-vmin * inv_s);

  int isum = 0;
  for (int i = tid; i < K; i += blockDim.x) {
    float v = __builtin_nontemporal_load(&s[i]);
    float q = rintf(v * inv_s) + zp;
    q = fminf(fmaxf(q, 0.0f), qmax);
    int qi = (int)q;
    dst[(size_t)row * K + i] = (uint8_t)qi;
    isum += qi;
  }
  for (int o = 16; o > 0; o >>= 1) isum += __shfl_xor(isum, o, WAVE);
  if (lane == 0) ssum[wave] = isum;
  __syncthreads();
  if (tid == 0) {
    int t = 0;
    for (int w = 0; w < nwaves; ++w) t += ssum[w];
    param[row] = float4{scale, zp, (float)t, 0.0f};
  }
}

__global__ __launch_bounds__(256, 1) void qgemm_wmma_kernel(
    const uint8_t* __restrict__ Xq, const uint8_t* __restrict__ Wq,
    const float4* __restrict__ xparam, const float4* __restrict__ wparam,
    const float* __restrict__ bias, float* __restrict__ out,
    int M, int N, int K) {
  const int lane = threadIdx.x & (WAVE - 1);
  const int wave = threadIdx.x >> 5;
  const int half = lane >> 4;
  const int l16  = lane & 15;
  const int wm   = wave & 3;
  const int wn   = wave >> 2;
  const int mBase = blockIdx.y * 128 + wm * 32;
  const int nBase = blockIdx.x * 128 + wn * 64;

  v8i acc[2][4] = {};

  uint32_t aOff[2];
  uint32_t bOff[4];
#pragma unroll
  for (int t = 0; t < 2; ++t)
    aOff[t] = (uint32_t)(mBase + t * 16 + l16) * (uint32_t)K + half * 8u;
#pragma unroll
  for (int u = 0; u < 4; ++u)
    bOff[u] = (uint32_t)(nBase + u * 16 + l16) * (uint32_t)K + half * 8u;

  const int ksteps = K >> 6;
  for (int ks = 0; ks < ksteps; ++ks) {
    const uint8_t* aB = Xq + ks * 64;
    const uint8_t* bB = Wq + ks * 64;

    __builtin_prefetch(aB + aOff[0] + 64, 0, 1);
    __builtin_prefetch(aB + aOff[1] + 64, 0, 1);

    v8i a[2], b[4];
#pragma unroll
    for (int t = 0; t < 2; ++t) {
      uint2 d0 = *(const uint2*)(aB + aOff[t]);
      uint2 d1 = *(const uint2*)(aB + aOff[t] + 16);
      uint2 d2 = *(const uint2*)(aB + aOff[t] + 32);
      uint2 d3 = *(const uint2*)(aB + aOff[t] + 48);
      a[t] = (v8i){(int)d0.x, (int)d0.y, (int)d1.x, (int)d1.y,
                   (int)d2.x, (int)d2.y, (int)d3.x, (int)d3.y};
    }
#pragma unroll
    for (int u = 0; u < 4; ++u) {
      uint2 e0 = *(const uint2*)(bB + bOff[u]);
      uint2 e1 = *(const uint2*)(bB + bOff[u] + 16);
      uint2 e2 = *(const uint2*)(bB + bOff[u] + 32);
      uint2 e3 = *(const uint2*)(bB + bOff[u] + 48);
      b[u] = (v8i){(int)e0.x, (int)e0.y, (int)e1.x, (int)e1.y,
                   (int)e2.x, (int)e2.y, (int)e3.x, (int)e3.y};
    }

#pragma unroll
    for (int t = 0; t < 2; ++t)
#pragma unroll
      for (int u = 0; u < 4; ++u)
        acc[t][u] = __builtin_amdgcn_wmma_i32_16x16x64_iu8(
             false, a[t],  false, b[u],
            acc[t][u],  false,  false);
  }

  const float fK = (float)K;
#pragma unroll
  for (int t = 0; t < 2; ++t) {
    const int m0 = mBase + t * 16 + half * 8;
    float4 xp[8];
#pragma unroll
    for (int r = 0; r < 8; ++r) xp[r] = xparam[m0 + r];
#pragma unroll
    for (int u = 0; u < 4; ++u) {
      const int n = nBase + u * 16 + l16;
      const float4 wp = wparam[n];
      const float bn = bias[n];
      float* o = out + (size_t)m0 * N + n;
#pragma unroll
      for (int r = 0; r < 8; ++r) {
        float s = (float)acc[t][u][r];
        float corr = s - wp.y * xp[r].z - xp[r].y * wp.z + fK * xp[r].y * wp.y;
        __builtin_nontemporal_store(xp[r].x * wp.x * corr + bn, &o[(size_t)r * N]);
      }
    }
  }
}

extern "C" void kernel_launch(void* const* d_in, const int* in_sizes, int n_in,
                              void* d_out, int out_size, void* d_ws, size_t ws_size,
                              hipStream_t stream) {
  const float* x    = (const float*)d_in[0];
  const float* w    = (const float*)d_in[1];
  const float* bias = (const float*)d_in[2];
  float* out        = (float*)d_out;

  const int N = in_sizes[2];
  const int K = in_sizes[1] / N;
  const int M = in_sizes[0] / K;

  uint8_t* ws     = (uint8_t*)d_ws;
  uint8_t* Wq     = ws;
  uint8_t* Xq     = ws + (size_t)N * K;
  float4*  wparam = (float4*)(Xq + (size_t)M * K);
  float4*  xparam = wparam + N;

  rowquant_kernel<<<N, 256, 0, stream>>>(w, Wq, wparam, K, 15.0f);
  rowquant_kernel<<<M, 256, 0, stream>>>(x, Xq, xparam, K, 255.0f);

  dim3 grid(N / 128, M / 128);
  qgemm_wmma_kernel<<<grid, 256, 0, stream>>>(Xq, Wq, xparam, wparam, bias, out,
                                              M, N, K);
}
